// enc_dec_model_120259084743
// MI455X (gfx1250) — matrix operand order repaired and checked by reading, not compiled, not run on hardware
//
#include <hip/hip_runtime.h>
#include <hip/hip_bf16.h>

// ---------------------------------------------------------------------------
// Seq2seq LSTM (B=64, Tx=512, F=64, H=512, OUT=64) tuned for MI455X (gfx1250).
//
//  * Encoder = 512 strictly sequential LSTM steps; each step is a
//    [64 x 576] @ [576 x 2048] GEMM (151 MFLOP) + elementwise gates.
//    Latency-bound: persistent 4-WG grid (32 waves), one global
//    release/acquire barrier per step.
//  * bf16 WMMA (v_wmma_f32_16x16x32_bf16, f32 accum); weights repacked ONCE
//    into the exact per-lane B-fragment VGPR layout (32B/lane contiguous) so
//    the inner loop is pure global_load_b128 + v_wmma.
//  * h|x_t double-buffered in a 144KB global bf16 tensor (L2-resident);
//    cell state c never leaves registers.
//  * Gate nonlinearities use the native gfx1250 V_TANH_F32 when the builtin
//    exists (sigmoid derived from tanh), else a single-__expf + v_rcp path —
//    the libm tanhf expansion dominated VALU count in the previous round.
// ---------------------------------------------------------------------------

typedef __attribute__((ext_vector_type(16))) __bf16 v16bf;
typedef __attribute__((ext_vector_type(8)))  float  v8f;

#define NWG        4
#define TPB        256
#define BDIM       64
#define FDIM       64
#define HDIM       512
#define TXLEN      512
#define NCOL       2048          // 4*H
#define KEXT       576           // H + F
#define NK_ENC     18            // 576/32
#define NK_DEC     16            // 512/32

// workspace layout (bytes)
#define WS_BAR     0
#define WS_HEXT    256                                // 2 * 64*576*2   = 147456
#define WS_BENC    (256 + 2*BDIM*KEXT*2)              // enc pack: 2359296
#define WS_BDEC    (WS_BENC + 128*NK_ENC*32*32)       // dec pack: 2097152
#define WS_HDEC    (WS_BDEC + 128*NK_DEC*32*32)       // h_dec f32: 131072

union FragU { v16bf v; uint4 q[2]; };

__device__ __forceinline__ unsigned short f2bf(float f) {
  unsigned u = __float_as_uint(f);
  u += 0x7FFFu + ((u >> 16) & 1u);          // round-to-nearest-even
  return (unsigned short)(u >> 16);
}

__device__ __forceinline__ float fast_rcp(float x) {
#if __has_builtin(__builtin_amdgcn_rcpf)
  return __builtin_amdgcn_rcpf(x);
#else
  return 1.0f / x;
#endif
}

__device__ __forceinline__ float fast_tanh(float x) {
#if __has_builtin(__builtin_amdgcn_tanhf)
  return __builtin_amdgcn_tanhf(x);          // native V_TANH_F32 on gfx1250
#else
  // tanh(x) = 1 - 2/(exp(2x)+1); single v_exp_f32 + v_rcp_f32
  return 1.0f - 2.0f * fast_rcp(__expf(2.0f * x) + 1.0f);
#endif
}
__device__ __forceinline__ float sigm(float x) {
  return 0.5f * fast_tanh(0.5f * x) + 0.5f;  // sigma(x) = (1+tanh(x/2))/2
}

__device__ __forceinline__ v8f zero8() {
  v8f z;
#pragma unroll
  for (int i = 0; i < 8; ++i) z[i] = 0.0f;
  return z;
}

// A fragment (16x32 bf16) from row-major bf16 activations [64 x KEXT].
__device__ __forceinline__ v16bf load_afrag(const unsigned short* __restrict__ act,
                                            int mtile, int k0, int lanelo, int hi) {
  FragU f;
  const unsigned short* p = act + (mtile * 16 + lanelo) * KEXT + k0 + hi * 16;
  f.q[0] = *(const uint4*)p;
  f.q[1] = *(const uint4*)(p + 8);
  return f.v;
}

// B fragment (32x16 bf16) from the packed weight stream: 32 bytes per lane.
__device__ __forceinline__ v16bf load_bfrag(const unsigned* __restrict__ bp,
                                            int fragIdx, int lane) {
  FragU f;
  const uint4* p = (const uint4*)(bp + ((long)fragIdx * 32 + lane) * 8);
  f.q[0] = p[0];
  f.q[1] = p[1];
  return f.v;
}

// grid-wide release/acquire barrier (monotonic counter, no reset races)
__device__ __forceinline__ void grid_barrier(unsigned* bar, unsigned target) {
  __threadfence();                              // release (agent scope)
  __syncthreads();
  if (threadIdx.x == 0) {
    atomicAdd(bar, 1u);
    while (__hip_atomic_load(bar, __ATOMIC_RELAXED, __HIP_MEMORY_SCOPE_AGENT) < target)
      __builtin_amdgcn_s_sleep(1);
    __threadfence();                            // acquire
  }
  __syncthreads();
}

__global__ void init_bar_kernel(unsigned* bar) { bar[0] = 0u; }

// Repack [Wk ; Wtail] (rows 0..kLimit-1 from Wk, rest from Wtail) into WMMA
// B-fragment order: dword i = ((tile*nK + ks)*32 + lane)*8 + v,
//   n = tile*16 + (lane&15),  k = ks*32 + (lane>=16 ? 16:0) + 2v.
__global__ void pack_wmma_b(const float* __restrict__ Wk,
                            const float* __restrict__ Wtail,
                            unsigned* __restrict__ dst, int nK, int kLimit) {
  long i = (long)blockIdx.x * blockDim.x + threadIdx.x;
  long total = (long)128 * nK * 32 * 8;
  if (i >= total) return;
  int v    = (int)(i & 7);
  int lane = (int)((i >> 3) & 31);
  long rem = i >> 8;                                  // tile*nK + ks
  int n = (int)((rem / nK) * 16) + (lane & 15);
  int k = (int)((rem % nK) * 32) + ((lane >> 4) ? 16 : 0) + 2 * v;
  float f0, f1;
  if (k < kLimit) {
    f0 = Wk[(long)k * NCOL + n];
    f1 = Wk[(long)(k + 1) * NCOL + n];
  } else {
    f0 = Wtail[(long)(k - kLimit) * NCOL + n];
    f1 = Wtail[(long)(k + 1 - kLimit) * NCOL + n];
  }
  dst[i] = (unsigned)f2bf(f0) | ((unsigned)f2bf(f1) << 16);
}

__global__ void __launch_bounds__(TPB, 1)
lstm_persist(const float* __restrict__ input_seq,
             const float* __restrict__ b_enc,
             const float* __restrict__ Wx_dec,
             const float* __restrict__ b_dec,
             const unsigned* __restrict__ Benc,
             const unsigned* __restrict__ Bdec,
             unsigned short* __restrict__ hext,   // [2][64][576] bf16
             float* __restrict__ hdec,            // [64][512] f32
             unsigned* __restrict__ bar) {
  const int tid    = threadIdx.x;
  const int wg     = blockIdx.x;
  const int lane   = tid & 31;
  const int wave   = tid >> 5;                 // 0..7
  const int j      = wg * 8 + wave;            // 0..31 : H-column tile
  const int lanelo = lane & 15;
  const int hi     = lane >> 4;
  const int col    = j * 16 + lanelo;          // owned H column
  const int gtid   = wg * TPB + tid;           // 0..1023

  // per-lane encoder biases (one per gate)
  const float bi = b_enc[0 * HDIM + col];
  const float bf_ = b_enc[1 * HDIM + col];
  const float bg = b_enc[2 * HDIM + col];
  const float bo = b_enc[3 * HDIM + col];

  // init buffer 0: h = 0, x-region = x_0
  for (int idx = gtid; idx < BDIM * KEXT; idx += NWG * TPB) {
    int row = idx / KEXT, c = idx - row * KEXT;
    unsigned short val = 0;
    if (c >= HDIM) val = f2bf(input_seq[((long)row * TXLEN) * FDIM + (c - HDIM)]);
    hext[idx] = val;
  }

  unsigned btarget = 0;
  grid_barrier(bar, (btarget += NWG));

  float c_st[4][8];
#pragma unroll
  for (int m = 0; m < 4; ++m)
#pragma unroll
    for (int r = 0; r < 8; ++r) c_st[m][r] = 0.0f;

  // ------------------------- encoder time loop --------------------------
  for (int t = 0; t < TXLEN; ++t) {
    const unsigned short* hin = hext + (t & 1) * (BDIM * KEXT);
    unsigned short* hout      = hext + ((t + 1) & 1) * (BDIM * KEXT);

    v8f acc[4][4];
#pragma unroll
    for (int m = 0; m < 4; ++m)
#pragma unroll
      for (int g = 0; g < 4; ++g) acc[m][g] = zero8();

    for (int ks = 0; ks < NK_ENC; ++ks) {
      v16bf bfrag[4];
#pragma unroll
      for (int g = 0; g < 4; ++g)
        bfrag[g] = load_bfrag(Benc, (g * 32 + j) * NK_ENC + ks, lane);
      if (ks + 1 < NK_ENC) {
#pragma unroll
        for (int g = 0; g < 4; ++g)
          __builtin_prefetch(Benc + ((long)((g * 32 + j) * NK_ENC + ks + 1) * 32 + lane) * 8, 0, 3);
      }
#pragma unroll
      for (int m = 0; m < 4; ++m) {
        v16bf a = load_afrag(hin, m, ks * 32, lanelo, hi);
#pragma unroll
        for (int g = 0; g < 4; ++g)
          acc[m][g] = __builtin_amdgcn_wmma_f32_16x16x32_bf16(
              false, a, false, bfrag[g], (short)0, acc[m][g], false, false);
      }
    }

    // gates -> c,h ; write next-step h (other buffer, no read/write overlap)
#pragma unroll
    for (int m = 0; m < 4; ++m) {
#pragma unroll
      for (int r = 0; r < 8; ++r) {
        float zi = acc[m][0][r] + bi;
        float zf = acc[m][1][r] + bf_;
        float zg = acc[m][2][r] + bg;
        float zo = acc[m][3][r] + bo;
        float ig = sigm(zi), fg = sigm(zf);
        float gg = fast_tanh(zg), og = sigm(zo);
        float cn = fg * c_st[m][r] + ig * gg;
        c_st[m][r] = cn;
        float hn = og * fast_tanh(cn);
        int row = m * 16 + hi * 8 + r;
        hout[row * KEXT + col] = f2bf(hn);
      }
    }
    // stage x_{t+1} into the next buffer (distributed over all 1024 threads)
    if (t + 1 < TXLEN) {
#pragma unroll
      for (int e = 0; e < (BDIM * FDIM) / (NWG * TPB); ++e) {
        int idx = gtid + e * (NWG * TPB);
        int row = idx >> 6, fcol = idx & 63;
        hout[row * KEXT + HDIM + fcol] =
            f2bf(input_seq[((long)row * TXLEN + (t + 1)) * FDIM + fcol]);
      }
    }
    grid_barrier(bar, (btarget += NWG));
  }

  // ---------------- decoder: one LSTM step, x = input[:, -1, -1] --------
  {
    const unsigned short* hin = hext;       // buffer 0 holds h after step 511
    v8f acc[4][4];
#pragma unroll
    for (int m = 0; m < 4; ++m)
#pragma unroll
      for (int g = 0; g < 4; ++g) acc[m][g] = zero8();

    for (int ks = 0; ks < NK_DEC; ++ks) {
      v16bf bfrag[4];
#pragma unroll
      for (int g = 0; g < 4; ++g)
        bfrag[g] = load_bfrag(Bdec, (g * 32 + j) * NK_DEC + ks, lane);
#pragma unroll
      for (int m = 0; m < 4; ++m) {
        v16bf a = load_afrag(hin, m, ks * 32, lanelo, hi);
#pragma unroll
        for (int g = 0; g < 4; ++g)
          acc[m][g] = __builtin_amdgcn_wmma_f32_16x16x32_bf16(
              false, a, false, bfrag[g], (short)0, acc[m][g], false, false);
      }
    }

    const float bdi = b_dec[0 * HDIM + col], bdf = b_dec[1 * HDIM + col];
    const float bdg = b_dec[2 * HDIM + col], bdo = b_dec[3 * HDIM + col];
    const float wxi = Wx_dec[0 * HDIM + col], wxf = Wx_dec[1 * HDIM + col];
    const float wxg = Wx_dec[2 * HDIM + col], wxo = Wx_dec[3 * HDIM + col];

#pragma unroll
    for (int m = 0; m < 4; ++m) {
#pragma unroll
      for (int r = 0; r < 8; ++r) {
        int row = m * 16 + hi * 8 + r;
        float y = input_seq[((long)row * TXLEN + (TXLEN - 1)) * FDIM + (FDIM - 1)];
        float zi = acc[m][0][r] + bdi + y * wxi;
        float zf = acc[m][1][r] + bdf + y * wxf;
        float zg = acc[m][2][r] + bdg + y * wxg;
        float zo = acc[m][3][r] + bdo + y * wxo;
        float ig = sigm(zi), fg = sigm(zf);
        float gg = fast_tanh(zg), og = sigm(zo);
        float cn = fg * c_st[m][r] + ig * gg;
        float hn = og * fast_tanh(cn);
        hdec[row * HDIM + col] = hn;
      }
    }
  }
}

// final dense: out[b, o] = b_out[o] + sum_k hdec[b,k] * W_out[k,o]   (64x64)
__global__ void proj_kernel(const float* __restrict__ hdec,
                            const float* __restrict__ W_out,
                            const float* __restrict__ b_out,
                            float* __restrict__ out) {
  int o = blockIdx.x * blockDim.x + threadIdx.x;   // 0..4095
  int b = o >> 6, oc = o & 63;
  float s = b_out[oc];
  const float* h = hdec + (long)b * HDIM;
  for (int k = 0; k < HDIM; ++k) s = fmaf(h[k], W_out[k * FDIM + oc], s);
  out[o] = s;
}

extern "C" void kernel_launch(void* const* d_in, const int* in_sizes, int n_in,
                              void* d_out, int out_size, void* d_ws, size_t ws_size,
                              hipStream_t stream) {
  const float* input_seq = (const float*)d_in[0];
  const float* Wx_enc    = (const float*)d_in[1];
  const float* Wh_enc    = (const float*)d_in[2];
  const float* b_enc     = (const float*)d_in[3];
  const float* Wx_dec    = (const float*)d_in[4];
  const float* Wh_dec    = (const float*)d_in[5];
  const float* b_dec     = (const float*)d_in[6];
  const float* W_out     = (const float*)d_in[7];
  const float* b_out     = (const float*)d_in[8];
  float* out = (float*)d_out;

  char* ws = (char*)d_ws;
  unsigned*       bar   = (unsigned*)(ws + WS_BAR);
  unsigned short* hext  = (unsigned short*)(ws + WS_HEXT);
  unsigned*       Benc  = (unsigned*)(ws + WS_BENC);
  unsigned*       Bdec  = (unsigned*)(ws + WS_BDEC);
  float*          hdecb = (float*)(ws + WS_HDEC);

  init_bar_kernel<<<1, 1, 0, stream>>>(bar);
  pack_wmma_b<<<(128 * NK_ENC * 32 * 8) / 256, 256, 0, stream>>>(
      Wh_enc, Wx_enc, Benc, NK_ENC, HDIM);
  pack_wmma_b<<<(128 * NK_DEC * 32 * 8) / 256, 256, 0, stream>>>(
      Wh_dec, Wx_dec, Bdec, NK_DEC, HDIM);   // k never reaches tail for dec
  lstm_persist<<<NWG, TPB, 0, stream>>>(
      input_seq, b_enc, Wx_dec, b_dec, Benc, Bdec, hext, hdecb, bar);
  proj_kernel<<<(BDIM * FDIM) / 256, 256, 0, stream>>>(hdecb, W_out, b_out, out);
}
